// GearNetTransformer_28321014350171
// MI455X (gfx1250) — compile-verified, matrix operand order checked by reading, not run on hardware
//
#include <hip/hip_runtime.h>
#include <hip/hip_bf16.h>

// ---------------------------------------------------------------------------
// GearNet (6-layer geometric relational GCN) for MI455X / gfx1250.
// All GEMMs via v_wmma_f32_16x16x32_bf16 (bf16 in, fp32 accum).
// K is zero-padded to a multiple of 32 in bf16 staging buffers and in the
// pre-transposed weights, so the WMMA K-loop is branchless and fragment loads
// are two b128 loads per lane. Relational scatter-add streams per relation so
// the fp32 accumulator (<= 33.5 MB) stays resident in the 192 MB L2.
// ---------------------------------------------------------------------------

typedef __attribute__((ext_vector_type(16))) __bf16       v16bf;
typedef __attribute__((ext_vector_type(8)))  float        v8f;
typedef __attribute__((ext_vector_type(4)))  unsigned int v4u;

union Frag { v4u u[2]; v16bf v; };

#define BN_EPS 1e-5f

// ---------------- utility kernels ----------------

__global__ void k_zero(float* __restrict__ p, long long n) {
  long long i = blockIdx.x * (long long)blockDim.x + threadIdx.x;
  if (i < n) p[i] = 0.f;
}

// fp32 [M,K] -> bf16 [M,Kp], zero-padding columns K..Kp-1
__global__ void k_f2bf_pad(const float* __restrict__ s, __bf16* __restrict__ d,
                           long long M, int K, int Kp) {
  long long i = blockIdx.x * (long long)blockDim.x + threadIdx.x;
  if (i >= M * Kp) return;
  long long m = i / Kp;
  int k = (int)(i % Kp);
  d[i] = (k < K) ? (__bf16)s[m * K + k] : (__bf16)0.f;
}

// fp32 weight [Ktot,N] row-major -> bf16 per-relation transposed blocks:
// dst[((r*N)+n)*Kp + k] where r = ksrc/Kblk, k = ksrc%Kblk. Padding pre-zeroed.
__global__ void k_wtp(const float* __restrict__ s, __bf16* __restrict__ d,
                      int Ktot, int N, int Kblk, int Kp) {
  long long i = blockIdx.x * (long long)blockDim.x + threadIdx.x;
  if (i >= (long long)Ktot * N) return;
  int ksrc = (int)(i / N), n = (int)(i % N);
  int r = ksrc / Kblk, k = ksrc % Kblk;
  d[((long long)r * N + n) * Kp + k] = (__bf16)s[i];
}

// predicated relational scatter-add: one block per edge
__global__ void k_scatter(const int* __restrict__ inidx, const int* __restrict__ outidx,
                          const int* __restrict__ rel, const float* __restrict__ w,
                          const float* __restrict__ src, float* __restrict__ dst,
                          int K, int r, int use_identity) {
  int e = blockIdx.x;
  if (rel[e] != r) return;
  int si = use_identity ? e : inidx[e];
  long long sb = (long long)si * K;
  long long db = (long long)outidx[e] * K;
  float we = w[e];
  for (int f = threadIdx.x; f < K; f += blockDim.x)
    atomicAdd(&dst[db + f], src[sb + f] * we);
}

// C[m*N+n] += b1[n] + b2[n]
__global__ void k_bias2(float* __restrict__ C, const float* __restrict__ b1,
                        const float* __restrict__ b2, long long M, int N) {
  long long i = blockIdx.x * (long long)blockDim.x + threadIdx.x;
  if (i >= M * N) return;
  int n = (int)(i % N);
  C[i] += b1[n] + b2[n];
}

// per-column mean / biased variance, one block per column
__global__ void k_stats(const float* __restrict__ X, int M, int N,
                        float* __restrict__ mean, float* __restrict__ var) {
  int n = blockIdx.x;
  float s = 0.f, s2 = 0.f;
  for (int m = threadIdx.x; m < M; m += blockDim.x) {
    float v = X[(long long)m * N + n];
    s += v; s2 += v * v;
  }
  __shared__ float sh[256], sh2[256];
  sh[threadIdx.x] = s; sh2[threadIdx.x] = s2;
  __syncthreads();
  for (int st = 128; st > 0; st >>= 1) {
    if (threadIdx.x < st) { sh[threadIdx.x] += sh[threadIdx.x + st]; sh2[threadIdx.x] += sh2[threadIdx.x + st]; }
    __syncthreads();
  }
  if (threadIdx.x == 0) {
    float mu = sh[0] / (float)M;
    mean[n] = mu;
    var[n]  = sh2[0] / (float)M - mu * mu;
  }
}

// y = bn(x) [-> relu] [+ shortcut]; optional second strided output (in-place safe)
__global__ void k_bn(const float* X, const float* mean, const float* var,
                     const float* g, const float* b, const float* shortcut,
                     float* out, float* out2, int ld2, long long M, int N, int do_relu) {
  long long i = blockIdx.x * (long long)blockDim.x + threadIdx.x;
  if (i >= M * N) return;
  int n = (int)(i % N);
  float v = (X[i] - mean[n]) * __frsqrt_rn(var[n] + BN_EPS) * g[n] + b[n];
  if (do_relu) v = fmaxf(v, 0.f);
  if (shortcut) v += shortcut[i];
  out[i] = v;
  if (out2) out2[(i / N) * (long long)ld2 + n] = v;
}

// H += relu(U + bias[n])
__global__ void k_addrelu(float* __restrict__ H, const float* __restrict__ U,
                          const float* __restrict__ bias, long long M, int N) {
  long long i = blockIdx.x * (long long)blockDim.x + threadIdx.x;
  if (i >= M * N) return;
  H[i] += fmaxf(U[i] + bias[(int)(i % N)], 0.f);
}

// graph pooling: gf[g, c] += nf[n, c]
__global__ void k_pool(const float* __restrict__ nf, const int* __restrict__ n2g,
                       float* __restrict__ gf, int NN, int D) {
  long long i = blockIdx.x * (long long)blockDim.x + threadIdx.x;
  if (i >= (long long)NN * D) return;
  int n = (int)(i / D), c = (int)(i % D);
  atomicAdd(&gf[(long long)n2g[n] * D + c], nf[i]);
}

// ---------------- WMMA GEMM ----------------
// C[M,N] (+)= A[M,Kp](bf16) * W, W supplied transposed+padded: WT[n*Kp + k].
// Kp is a multiple of 32, M a multiple of 32. 8 waves/block; each wave owns a
// 32x64 C tile (2 A frags x 4 B frags -> 8 WMMAs per K-step, branchless loop,
// fragments loaded as 2x b128 per lane per CDNA5 7.12.2 layout).
__global__ __launch_bounds__(256) void k_gemm(
    const __bf16* __restrict__ A,
    const __bf16* __restrict__ WT,
    float* __restrict__ C, int ldc,
    int M, int N, int Kp, int acc) {
  int wave = threadIdx.x >> 5;
  int lane = threadIdx.x & 31;
  int half = lane >> 4;     // K sub-block select for A/B; +8 row offset for C/D
  int l16  = lane & 15;     // A: row in tile; B/C: column in tile
  int m0 = (blockIdx.y * 8 + wave) * 32;
  int n0 = blockIdx.x * 64;
  if (m0 >= M) return;

  v8f c[2][4];
#pragma unroll
  for (int f = 0; f < 2; ++f)
#pragma unroll
    for (int t = 0; t < 4; ++t)
#pragma unroll
      for (int v = 0; v < 8; ++v) {
        int col = n0 + t * 16 + l16;
        c[f][t][v] = (acc && col < N)
                   ? C[(long long)(m0 + f * 16 + v + half * 8) * ldc + col] : 0.f;
      }

  const __bf16* a0 = A + (long long)(m0 + l16) * Kp + 8 * half;
  const __bf16* a1 = a0 + 16LL * Kp;
  const __bf16* bp[4];
#pragma unroll
  for (int t = 0; t < 4; ++t) {
    int col = n0 + t * 16 + l16;
    if (col >= N) col = N - 1;  // clamp: extra columns never stored
    bp[t] = WT + (long long)col * Kp + 8 * half;
  }

  for (int k0 = 0; k0 < Kp; k0 += 32) {
    Frag fa0, fa1;
    fa0.u[0] = *(const v4u*)(a0 + k0);
    fa0.u[1] = *(const v4u*)(a0 + k0 + 16);
    fa1.u[0] = *(const v4u*)(a1 + k0);
    fa1.u[1] = *(const v4u*)(a1 + k0 + 16);
#pragma unroll
    for (int t = 0; t < 4; ++t) {
      Frag fb;
      fb.u[0] = *(const v4u*)(bp[t] + k0);
      fb.u[1] = *(const v4u*)(bp[t] + k0 + 16);
      c[0][t] = __builtin_amdgcn_wmma_f32_16x16x32_bf16(
          false, fa0.v, false, fb.v, (short)0, c[0][t], false, false);
      c[1][t] = __builtin_amdgcn_wmma_f32_16x16x32_bf16(
          false, fa1.v, false, fb.v, (short)0, c[1][t], false, false);
    }
  }

#pragma unroll
  for (int f = 0; f < 2; ++f)
#pragma unroll
    for (int t = 0; t < 4; ++t)
#pragma unroll
      for (int v = 0; v < 8; ++v) {
        int col = n0 + t * 16 + l16;
        if (col < N)
          C[(long long)(m0 + f * 16 + v + half * 8) * ldc + col] = c[f][t][v];
      }
}

// ---------------- host orchestration ----------------

static inline long long cdiv(long long a, long long b) { return (a + b - 1) / b; }
static inline int padK(int k) { return (k + 31) & ~31; }

extern "C" void kernel_launch(void* const* d_in, const int* in_sizes, int n_in,
                              void* d_out, int out_size, void* d_ws, size_t ws_size,
                              hipStream_t stream) {
  (void)in_sizes; (void)out_size; (void)ws_size;

  const int N_NODE = 2048, N_EDGE = 16384, N_LINE = 65536;
  const int dims[7] = {21, 512, 512, 512, 512, 512, 512};
  int din[6], ein[6], eout[6];
  for (int i = 0; i < 6; ++i) {
    din[i]  = dims[i];
    ein[i]  = (i == 0) ? 59 : dims[i - 1];
    eout[i] = dims[i];
  }
  const int dO = 512, FEAT = 3072;

  // ---- inputs (setup_inputs dict order; params flattened via sorted keys) ----
  const float* x        = (const float*)d_in[0];
  const float* edge_fea = (const float*)d_in[1];
  const float* ew       = (const float*)d_in[2];
  const float* lew      = (const float*)d_in[3];
  const int PB = 4;
  // edge layer leaves: W_lin, W_sl, b_lin, b_sl, bn_b, bn_g
  auto ELp = [&](int i, int j) { return (const float*)d_in[PB + 6 * i + j]; };
  // node layer leaves: W_lin, W_sl, b_lin, b_sl, bn2_b, bn2_g, bn_b, bn_g
  auto NLp = [&](int i, int j) { return (const float*)d_in[PB + 36 + 8 * i + j]; };
  const int* e_in  = (const int*)d_in[n_in - 7];
  const int* e_out = (const int*)d_in[n_in - 6];
  const int* e_rel = (const int*)d_in[n_in - 5];
  const int* l_in  = (const int*)d_in[n_in - 4];
  const int* l_out = (const int*)d_in[n_in - 3];
  const int* l_rel = (const int*)d_in[n_in - 2];
  const int* n2g   = (const int*)d_in[n_in - 1];

  // ---- workspace carving ----
  char* ws = (char*)d_ws;
  size_t off = 0;
  auto carve = [&](size_t bytes) -> char* {
    char* p = ws + off;
    off += (bytes + 255) & ~(size_t)255;
    return p;
  };
  size_t w0 = off;
  __bf16 *nWlinT[6], *nWslT[6], *eWlinT[6], *eWslT[6];
  for (int i = 0; i < 6; ++i) {
    int KpN = padK(din[i]), KpE = padK(ein[i]);
    nWlinT[i] = (__bf16*)carve(2ull * 7 * dO * KpN);        // [7][512][KpN]
    nWslT[i]  = (__bf16*)carve(2ull * dO * KpN);            // [512][KpN]
    eWlinT[i] = (__bf16*)carve(2ull * 8 * eout[i] * KpE);   // [8][eout][KpE]
    eWslT[i]  = (__bf16*)carve(2ull * eout[i] * KpE);       // [eout][KpE]
  }
  size_t w1 = off;
  float*  msg   = (float*)carve(4ull * N_EDGE * 512);
  __bf16* msgb  = (__bf16*)carve(2ull * N_EDGE * 512);
  __bf16* actb  = (__bf16*)carve(2ull * N_EDGE * 512);
  float*  Hn    = (float*)carve(4ull * N_NODE * 512);
  float*  Un    = (float*)carve(4ull * N_NODE * 512);
  float*  nodeA = (float*)carve(4ull * N_NODE * 512);
  float*  nodeB = (float*)carve(4ull * N_NODE * 512);
  float*  He    = (float*)carve(4ull * N_EDGE * 512);
  float*  edgeA = (float*)carve(4ull * N_EDGE * 512);
  float*  edgeB = (float*)carve(4ull * N_EDGE * 512);
  float*  mean  = (float*)carve(4ull * 512);
  float*  var   = (float*)carve(4ull * 512);

  // ---- launch helpers ----
  auto zero = [&](float* p, long long n) {
    k_zero<<<dim3((unsigned)cdiv(n, 256)), dim3(256), 0, stream>>>(p, n);
  };
  auto cvt = [&](const float* s, __bf16* d, long long M, int K, int Kp) {
    long long n = M * Kp;
    k_f2bf_pad<<<dim3((unsigned)cdiv(n, 256)), dim3(256), 0, stream>>>(s, d, M, K, Kp);
  };
  auto wtp = [&](const float* s, __bf16* d, int Ktot, int N, int Kblk, int Kp) {
    long long n = (long long)Ktot * N;
    k_wtp<<<dim3((unsigned)cdiv(n, 256)), dim3(256), 0, stream>>>(s, d, Ktot, N, Kblk, Kp);
  };
  auto gemm = [&](const __bf16* A, const __bf16* WT, float* C, int ldc,
                  int M, int N, int Kp, int acc) {
    dim3 g((unsigned)cdiv(N, 64), (unsigned)cdiv(M, 256));
    k_gemm<<<g, dim3(256), 0, stream>>>(A, WT, C, ldc, M, N, Kp, acc);
  };
  auto stats = [&](const float* X, int M, int N) {
    k_stats<<<dim3(N), dim3(256), 0, stream>>>(X, M, N, mean, var);
  };

  // ---- one-time weight prep: zero padded region, convert+transpose to bf16 ----
  zero((float*)(ws + w0), (long long)(w1 - w0) / 4);
  for (int i = 0; i < 6; ++i) {
    int KpN = padK(din[i]), KpE = padK(ein[i]);
    wtp(NLp(i, 0), nWlinT[i], 7 * din[i], dO, din[i], KpN);
    wtp(NLp(i, 1), nWslT[i], din[i], dO, din[i], KpN);
    wtp(ELp(i, 0), eWlinT[i], 8 * ein[i], eout[i], ein[i], KpE);
    wtp(ELp(i, 1), eWslT[i], ein[i], eout[i], ein[i], KpE);
  }

  float* gf = (float*)d_out;          // [8, 3072]
  float* nf = gf + 8 * FEAT;          // [2048, 3072]
  const float* nin  = x;              // current node input
  const float* einp = edge_fea;       // current edge (line-graph node) input

  for (int i = 0; i < 6; ++i) {
    const int dI = din[i], eI = ein[i], eO = eout[i];  // note: eO == dI
    const int KpN = padK(dI), KpE = padK(eI);

    // ---- node RGCN: Hn = relu(bn(upd@W_lin + x@W_sl + b)) [+ shortcut] ----
    cvt(nin, actb, N_NODE, dI, KpN);
    gemm(actb, nWslT[i], Hn, dO, N_NODE, dO, KpN, 0);
    for (int r = 0; r < 7; ++r) {
      zero(msg, (long long)N_NODE * dI);
      k_scatter<<<dim3(N_EDGE), dim3(256), 0, stream>>>(e_in, e_out, e_rel, ew, nin, msg, dI, r, 0);
      cvt(msg, msgb, N_NODE, dI, KpN);
      gemm(msgb, nWlinT[i] + (long long)r * dO * KpN, Hn, dO, N_NODE, dO, KpN, 1);
    }
    k_bias2<<<dim3((unsigned)cdiv((long long)N_NODE * dO, 256)), dim3(256), 0, stream>>>(
        Hn, NLp(i, 2), NLp(i, 3), N_NODE, dO);
    stats(Hn, N_NODE, dO);
    k_bn<<<dim3((unsigned)cdiv((long long)N_NODE * dO, 256)), dim3(256), 0, stream>>>(
        Hn, mean, var, NLp(i, 7), NLp(i, 6), (dI == dO) ? nin : nullptr,
        Hn, nullptr, 0, N_NODE, dO, 1);

    // ---- edge RGCN on the line graph ----
    cvt(einp, actb, N_EDGE, eI, KpE);
    gemm(actb, eWslT[i], He, eO, N_EDGE, eO, KpE, 0);
    for (int r = 0; r < 8; ++r) {
      zero(msg, (long long)N_EDGE * eI);
      k_scatter<<<dim3(N_LINE), dim3(256), 0, stream>>>(l_in, l_out, l_rel, lew, einp, msg, eI, r, 0);
      cvt(msg, msgb, N_EDGE, eI, KpE);
      gemm(msgb, eWlinT[i] + (long long)r * eO * KpE, He, eO, N_EDGE, eO, KpE, 1);
    }
    k_bias2<<<dim3((unsigned)cdiv((long long)N_EDGE * eO, 256)), dim3(256), 0, stream>>>(
        He, ELp(i, 2), ELp(i, 3), N_EDGE, eO);
    stats(He, N_EDGE, eO);
    float* ehid = (i & 1) ? edgeB : edgeA;
    k_bn<<<dim3((unsigned)cdiv((long long)N_EDGE * eO, 256)), dim3(256), 0, stream>>>(
        He, mean, var, ELp(i, 5), ELp(i, 4), nullptr, ehid, nullptr, 0, N_EDGE, eO, 1);

    // ---- edge-conditioned node update: Hn += relu(scatter(ehid*ew)@W_lin + b_lin) ----
    for (int r = 0; r < 7; ++r) {
      zero(msg, (long long)N_NODE * eO);
      k_scatter<<<dim3(N_EDGE), dim3(256), 0, stream>>>(nullptr, e_out, e_rel, ew, ehid, msg, eO, r, 1);
      cvt(msg, msgb, N_NODE, eO, KpN);
      gemm(msgb, nWlinT[i] + (long long)r * dO * KpN, Un, dO, N_NODE, dO, KpN, r > 0);
    }
    k_addrelu<<<dim3((unsigned)cdiv((long long)N_NODE * dO, 256)), dim3(256), 0, stream>>>(
        Hn, Un, NLp(i, 2), N_NODE, dO);

    // ---- outer BN (no relu); emit layer output + node_feature slice ----
    stats(Hn, N_NODE, dO);
    float* outN = (i & 1) ? nodeB : nodeA;
    k_bn<<<dim3((unsigned)cdiv((long long)N_NODE * dO, 256)), dim3(256), 0, stream>>>(
        Hn, mean, var, NLp(i, 5), NLp(i, 4), nullptr, outN, nf + i * dO, FEAT, N_NODE, dO, 0);

    nin  = outN;
    einp = ehid;
  }

  // ---- graph pooling ----
  zero(gf, 8LL * FEAT);
  k_pool<<<dim3((unsigned)cdiv((long long)N_NODE * FEAT, 256)), dim3(256), 0, stream>>>(
      nf, n2g, gf, N_NODE, FEAT);
}
